// MultiHead_94489281378
// MI455X (gfx1250) — matrix operand order repaired and checked by reading, not compiled, not run on hardware
//
#include <hip/hip_runtime.h>
#include <hip/hip_bf16.h>

// ---------------------------------------------------------------------------
// Multi-head causal attention, bf16 WMMA path for gfx1250 (MI455X).
// B=2, S=2048, D=1024, H=16, DH=64. SCALE=32, mask = -1e10 before /32.
//
// Pipeline:
//   1) convert: fp32 -> bf16 for q/k/v inputs and the 4 weight matrices
//   2) 3x projection GEMM (X @ W^T) via v_wmma_f32_16x16x32_bf16 with the
//      W tile staged in LDS per block through double-buffered async
//      global->LDS copies (global_load_async_to_lds_b128 / s_wait_asynccnt),
//      removing the 4x redundant per-wave W reads.
//      Q/K stored head-major [B,H,S,64], V stored transposed [B,H,64,S]
//   3) flash attention (causal, online softmax), one wave per 16-row q-tile,
//      row reductions via v_permlane16_b32 butterflies,
//      P transposed D-layout -> A-layout through per-wave LDS slice
//   4) output GEMM (O @ Wo^T) -> fp32 d_out
// ---------------------------------------------------------------------------

#define B_  2
#define S_  2048
#define D_  1024
#define H_  16
#define DH_ 64
#define M_  (B_ * S_)   // 4096 rows

typedef __attribute__((ext_vector_type(16))) __bf16 v16bf;
typedef __attribute__((ext_vector_type(8)))  __bf16 v8bf;
typedef __attribute__((ext_vector_type(8)))  float  v8f;
typedef __attribute__((ext_vector_type(4)))  int    v4i;

#define AS_GLOBAL __attribute__((address_space(1)))
#define AS_LDS    __attribute__((address_space(3)))

// ---------------------------------------------------------------------------
// Async global -> LDS copy (ASYNCcnt-tracked), guarded for toolchain support.
// Probe-derived signature: (addrspace(1) v4i*, addrspace(3) v4i*, i32, i32)
// ---------------------------------------------------------------------------
#if __has_builtin(__builtin_amdgcn_global_load_async_to_lds_b128)
#define HAVE_ASYNC_LDS 1
__device__ __forceinline__ void async_copy_b128(const __bf16* g, __bf16* l) {
  __builtin_amdgcn_global_load_async_to_lds_b128(
      (AS_GLOBAL v4i*)(void*)g, (AS_LDS v4i*)(void*)l, 0, 0);
}
__device__ __forceinline__ void wait_async_lds() {
#if __has_builtin(__builtin_amdgcn_s_wait_asynccnt)
  __builtin_amdgcn_s_wait_asynccnt(0);
#else
  asm volatile("s_wait_asynccnt 0x0" ::: "memory");
#endif
}
#else
#define HAVE_ASYNC_LDS 0
__device__ __forceinline__ void async_copy_b128(const __bf16* g, __bf16* l) {
  *(v8bf*)l = *(const v8bf*)g;   // synchronous fallback
}
__device__ __forceinline__ void wait_async_lds() {}
#endif

// ---------------------------------------------------------------------------
// Fragment loaders (wave32 WMMA; A/B in this code's contiguous k order, C/D per CDNA5 ISA 7.12.2)
//
// A-matrix 16x32 (16-bit): lane l (l&15 = M row), half = l>>4:
//   elements 0..15 -> K = 32*step + 16*half + i   (contiguous, matches B)
// B-matrix 32x16 (16-bit): lane l (l&15 = N col), half = l>>4:
//   elements 0..15 -> K = 32*step + 16*half + i   (contiguous 32 bytes)
// C/D 16x16 f32: lane l col N=l&15; VGPR r row M = r + 8*(l>>4)
// ---------------------------------------------------------------------------

__device__ __forceinline__ v16bf load_a_frag(const __bf16* __restrict__ X, int ld,
                                             int row, int k0, int lane) {
  const int kb = k0 + ((lane >> 4) << 4);
  const __bf16* p = X + (size_t)row * ld + kb;
  v8bf a0 = *(const v8bf*)p;        // 16B contiguous
  v8bf a1 = *(const v8bf*)(p + 8);  // 16B contiguous
  v16bf r;
#pragma unroll
  for (int i = 0; i < 8; ++i) { r[i] = a0[i]; r[i + 8] = a1[i]; }
  return r;
}

// B fragment from bf16 row-major matrix (rows = N, cols = K), global memory
__device__ __forceinline__ v16bf load_b_frag(const __bf16* __restrict__ Bm, int ld,
                                             int row, int k0, int lane) {
  const int kb = k0 + ((lane >> 4) << 4);
  return *(const v16bf*)(Bm + (size_t)row * ld + kb);  // 32B contiguous
}

// B fragment from LDS stage: rows padded to 40 elements (80B) to spread banks
#define BSTRIDE 40
__device__ __forceinline__ v16bf load_b_lds(const __bf16* base, int row, int half) {
  const __bf16* p = base + row * BSTRIDE + (half << 4);
  v8bf b0 = *(const v8bf*)p;        // 16B aligned
  v8bf b1 = *(const v8bf*)(p + 8);
  v16bf r;
#pragma unroll
  for (int i = 0; i < 8; ++i) { r[i] = b0[i]; r[i + 8] = b1[i]; }
  return r;
}

__device__ __forceinline__ v8f wmma_bf16(v16bf a, v16bf b, v8f c) {
  return __builtin_amdgcn_wmma_f32_16x16x32_bf16(
      /*neg_a=*/false, a, /*neg_b=*/false, b,
      /*c_mod=*/(short)0, c, /*reuse_a=*/false, /*reuse_b=*/false);
}

// ---------------------------------------------------------------------------
// 16-lane butterfly reductions (within each half-wave "row").
// v_permlane16_b32: lanes 0-7 select from s_lo nibbles, 8-15 from s_hi,
// applied independently per 16-lane row. Tables encode sel[l] = l ^ k.
// ---------------------------------------------------------------------------
#if __has_builtin(__builtin_amdgcn_permlane16)
__device__ __forceinline__ float xperm16(float v, unsigned lo, unsigned hi) {
  unsigned u = __float_as_uint(v);
  unsigned r = __builtin_amdgcn_permlane16(u, u, lo, hi, false, false);
  return __uint_as_float(r);
}
__device__ __forceinline__ float red_max16(float v) {
  v = fmaxf(v, xperm16(v, 0x67452301u, 0xEFCDAB89u));  // xor 1
  v = fmaxf(v, xperm16(v, 0x54761032u, 0xDCFE98BAu));  // xor 2
  v = fmaxf(v, xperm16(v, 0x32107654u, 0xBA98FEDCu));  // xor 4
  v = fmaxf(v, xperm16(v, 0xFEDCBA98u, 0x76543210u));  // xor 8
  return v;
}
__device__ __forceinline__ float red_sum16(float v) {
  v += xperm16(v, 0x67452301u, 0xEFCDAB89u);
  v += xperm16(v, 0x54761032u, 0xDCFE98BAu);
  v += xperm16(v, 0x32107654u, 0xBA98FEDCu);
  v += xperm16(v, 0xFEDCBA98u, 0x76543210u);
  return v;
}
#else
__device__ __forceinline__ float red_max16(float v) {
#pragma unroll
  for (int off = 1; off < 16; off <<= 1) v = fmaxf(v, __shfl_xor(v, off, 32));
  return v;
}
__device__ __forceinline__ float red_sum16(float v) {
#pragma unroll
  for (int off = 1; off < 16; off <<= 1) v += __shfl_xor(v, off, 32);
  return v;
}
#endif

// ---------------------------------------------------------------------------
// fp32 -> bf16 conversion pass (8 elements / thread, packed cvt + b128 store)
// ---------------------------------------------------------------------------
__global__ __launch_bounds__(256) void cvt_bf16_kernel(
    const float* __restrict__ src, __bf16* __restrict__ dst, int n) {
  const int i = (blockIdx.x * 256 + threadIdx.x) * 8;
  if (i < n) {
    v8f x = *(const v8f*)(src + i);
    v8bf y;
#pragma unroll
    for (int j = 0; j < 8; ++j) y[j] = (__bf16)x[j];
    *(v8bf*)(dst + i) = y;
  }
}

// ---------------------------------------------------------------------------
// GEMM: out[m,n] = sum_k X[m,k] * W[n,k]   (i.e. X @ W^T), all-bf16 operands.
// Block = 128 threads (4 waves); wave w -> 16-row m-tile; 64 cols per block.
// W tile (64 N x 32 K) staged in LDS via double-buffered async copies and
// shared by all 4 waves (4x L2 read reduction on the dominant stream).
// MODE 0: bf16 out, head-major  [B,H,S,DH]   (for Q, K)
// MODE 1: bf16 out, transposed  [B,H,DH,S]   (for V)
// MODE 2: fp32 out, row-major   [M,D]        (final output)
// ---------------------------------------------------------------------------
template <int MODE>
__global__ __launch_bounds__(128) void gemm_xwT_kernel(
    const __bf16* __restrict__ X, const __bf16* __restrict__ W,
    __bf16* __restrict__ outb, float* __restrict__ outf) {
  __shared__ __align__(128) __bf16 ldsB[2][64 * BSTRIDE];  // 2 x 5 KB

  const int tid  = threadIdx.x;
  const int lane = tid & 31;
  const int wave = tid >> 5;
  const int mt   = blockIdx.x * 4 + wave;   // 16-row tile index
  const int nb   = blockIdx.y * 64;         // column base
  const int col  = lane & 15;
  const int half = lane >> 4;
  const int arow = mt * 16 + col;

  // cooperative stage mapping: thread copies 32B of row (nb + tid/2)
  const int sn = tid >> 1;
  const int sk = (tid & 1) << 4;

  // prologue: kick k-step 0 into buffer 0
  {
    const __bf16* g = W + (size_t)(nb + sn) * D_ + sk;
    __bf16* l = &ldsB[0][sn * BSTRIDE + sk];
    async_copy_b128(g, l);
    async_copy_b128(g + 8, l + 8);
  }

  v8f acc[4] = {};
  const int NSTEP = D_ / 32;
  for (int step = 0; step < NSTEP; ++step) {
    const int k0 = step * 32;
    wait_async_lds();     // this wave's fills for buf[step&1] complete
    __syncthreads();      // all waves' fills complete; prior reads retired

    if (step + 1 < NSTEP) {  // overlap next tile DMA with this tile's WMMAs
      const __bf16* g = W + (size_t)(nb + sn) * D_ + (k0 + 32) + sk;
      __bf16* l = &ldsB[(step + 1) & 1][sn * BSTRIDE + sk];
      async_copy_b128(g, l);
      async_copy_b128(g + 8, l + 8);
    }

    v16bf a = load_a_frag(X, D_, arow, k0, lane);
    const __bf16* bbuf = ldsB[step & 1];
#pragma unroll
    for (int j = 0; j < 4; ++j) {
      v16bf b = load_b_lds(bbuf, j * 16 + col, half);
      acc[j] = wmma_bf16(a, b, acc[j]);
    }
  }

#pragma unroll
  for (int r = 0; r < 8; ++r) {
    const int mrow = mt * 16 + r + 8 * half;
    const int bb = mrow >> 11;        // / S_
    const int ss = mrow & (S_ - 1);
#pragma unroll
    for (int j = 0; j < 4; ++j) {
      const int ncol = nb + j * 16 + col;
      const float v = acc[j][r];
      if constexpr (MODE == 0) {
        const int h = ncol >> 6, d = ncol & 63;
        outb[(((size_t)bb * H_ + h) * S_ + ss) * DH_ + d] = (__bf16)v;
      } else if constexpr (MODE == 1) {
        const int h = ncol >> 6, d = ncol & 63;
        outb[(((size_t)bb * H_ + h) * DH_ + d) * S_ + ss] = (__bf16)v;
      } else {
        outf[(size_t)mrow * D_ + ncol] = v;
      }
    }
  }
}

// ---------------------------------------------------------------------------
// Flash attention (causal, online softmax). One wave per 16-row q-tile.
// logits = (q.k - 1e10*mask) / 32 ; masked entries set to -1e30 directly.
// ---------------------------------------------------------------------------
__global__ __launch_bounds__(128) void flash_attn_kernel(
    const __bf16* __restrict__ Qh,   // [B,H,S,DH]
    const __bf16* __restrict__ Kh,   // [B,H,S,DH]
    const __bf16* __restrict__ Vt,   // [B,H,DH,S]
    __bf16* __restrict__ O) {        // [B*S, D]
  __shared__ __align__(64) __bf16 ldsP[4][16 * 32];  // 1KB per wave

  const int lane = threadIdx.x & 31;
  const int wave = threadIdx.x >> 5;
  const int qt   = blockIdx.x * 4 + wave;  // q tile: 0..127
  const int bh   = blockIdx.y;             // 0..31
  const int bidx = bh >> 4;
  const int h    = bh & 15;
  const int col  = lane & 15;
  const int half = lane >> 4;

  const __bf16* Qb = Qh + (size_t)bh * S_ * DH_;
  const __bf16* Kb = Kh + (size_t)bh * S_ * DH_;
  const __bf16* Vb = Vt + (size_t)bh * DH_ * S_;
  __bf16* lp = ldsP[wave];

  // Q A-fragments for d=[0,32) and d=[32,64)
  const int qrow = qt * 16 + col;
  v16bf aq0 = load_a_frag(Qb, DH_, qrow, 0, lane);
  v16bf aq1 = load_a_frag(Qb, DH_, qrow, 32, lane);

  v8f o[4] = {};
  float mr[8], lr[8];
#pragma unroll
  for (int r = 0; r < 8; ++r) { mr[r] = -1e30f; lr[r] = 0.f; }

  const float kLog2e = 1.44269504088896340736f;
  const float kInvScale = 1.0f / 32.0f;
  const int nchunks = qt / 2 + 1;  // 32-key chunks up to (and incl.) diagonal

  for (int c = 0; c < nchunks; ++c) {
    const int kbase = c * 32;

    if (c + 1 < nchunks) {  // prefetch next K/V chunk rows (global_prefetch_b8)
      __builtin_prefetch(Kb + (size_t)(kbase + 32 + col) * DH_, 0, 1);
      __builtin_prefetch(Vb + (size_t)col * S_ + kbase + 32, 0, 1);
    }

    // ---- scores: Q(16x64) @ K^T(64x32) -> two 16x16 tiles -----------------
    v16bf bk00 = load_b_frag(Kb, DH_, kbase + col, 0, lane);
    v16bf bk01 = load_b_frag(Kb, DH_, kbase + col, 32, lane);
    v16bf bk10 = load_b_frag(Kb, DH_, kbase + 16 + col, 0, lane);
    v16bf bk11 = load_b_frag(Kb, DH_, kbase + 16 + col, 32, lane);
    v8f s0 = {}, s1 = {};
    s0 = wmma_bf16(aq0, bk00, s0);
    s0 = wmma_bf16(aq1, bk01, s0);
    s1 = wmma_bf16(aq0, bk10, s1);
    s1 = wmma_bf16(aq1, bk11, s1);

    // ---- online softmax ---------------------------------------------------
    const bool dom = (c == nchunks - 1);  // only diagonal chunk needs mask
    float p0[8], p1[8];
#pragma unroll
    for (int r = 0; r < 8; ++r) {
      const int row = qt * 16 + r + 8 * half;
      float t0 = s0[r] * kInvScale;
      float t1 = s1[r] * kInvScale;
      if (dom) {
        if (kbase + col > row)      t0 = -1e30f;
        if (kbase + 16 + col > row) t1 = -1e30f;
      }
      const float cm = red_max16(fmaxf(t0, t1));   // row max over 32 key cols
      const float mn = fmaxf(mr[r], cm);
      const float alpha = exp2f((mr[r] - mn) * kLog2e);
      mr[r] = mn;
      p0[r] = exp2f((t0 - mn) * kLog2e);
      p1[r] = exp2f((t1 - mn) * kLog2e);
      lr[r] = lr[r] * alpha + red_sum16(p0[r] + p1[r]);
      o[0][r] *= alpha; o[1][r] *= alpha; o[2][r] *= alpha; o[3][r] *= alpha;
    }

    // ---- transpose P (C/D layout -> A layout) through per-wave LDS --------
#pragma unroll
    for (int r = 0; r < 8; ++r) {
      const int prow = r + 8 * half;
      lp[prow * 32 + col]      = (__bf16)p0[r];
      lp[prow * 32 + 16 + col] = (__bf16)p1[r];
    }
    asm volatile("s_wait_dscnt 0x0" ::: "memory");  // LDS stores -> loads (intra-wave)
    v16bf pa = load_a_frag(lp, 32, col, 0, lane);

    // ---- O += P(16x32) @ V(32x64)  (V transposed: contiguous in key) ------
#pragma unroll
    for (int j = 0; j < 4; ++j) {
      v16bf bv = load_b_frag(Vb, S_, j * 16 + col, kbase, lane);
      o[j] = wmma_bf16(pa, bv, o[j]);
    }
  }

  // ---- epilogue: normalize and store bf16 [B*S, D] ------------------------
#pragma unroll
  for (int r = 0; r < 8; ++r) {
    const float inv = 1.0f / lr[r];
    const int srow = qt * 16 + r + 8 * half;
    const size_t mrow = (size_t)bidx * S_ + srow;
#pragma unroll
    for (int j = 0; j < 4; ++j) {
      O[mrow * D_ + h * 64 + j * 16 + col] = (__bf16)(o[j][r] * inv);
    }
  }
}

// ---------------------------------------------------------------------------
// Launch. Workspace (bf16 elements):
//   Xq|Xk|Xv (3 * 4M) | Wq|Wk|Wv|Wo (4 * 1M) | Qh|Kh|Vt|Ob (4 * 4M)
//   = 32M elements = 64 MB
// ---------------------------------------------------------------------------
extern "C" void kernel_launch(void* const* d_in, const int* in_sizes, int n_in,
                              void* d_out, int out_size, void* d_ws, size_t ws_size,
                              hipStream_t stream) {
  (void)in_sizes; (void)n_in; (void)out_size; (void)ws_size;
  const float* q  = (const float*)d_in[0];
  const float* k  = (const float*)d_in[1];
  const float* v  = (const float*)d_in[2];
  const float* Wq = (const float*)d_in[3];
  const float* Wk = (const float*)d_in[4];
  const float* Wv = (const float*)d_in[5];
  const float* Wo = (const float*)d_in[6];
  float* out = (float*)d_out;

  const size_t NE = (size_t)M_ * D_;   // 4194304
  const size_t NW = (size_t)D_ * D_;   // 1048576
  __bf16* Xq  = (__bf16*)d_ws;
  __bf16* Xk  = Xq + NE;
  __bf16* Xv  = Xk + NE;
  __bf16* Wqb = Xv + NE;
  __bf16* Wkb = Wqb + NW;
  __bf16* Wvb = Wkb + NW;
  __bf16* Wob = Wvb + NW;
  __bf16* Qh  = Wob + NW;
  __bf16* Kh  = Qh + NE;
  __bf16* Vt  = Kh + NE;
  __bf16* Ob  = Vt + NE;

  const int cb_x = (int)(NE / 8 / 256);  // 2048 blocks per 4M-element tensor
  const int cb_w = (int)(NW / 8 / 256);  // 512 blocks per 1M-element tensor
  cvt_bf16_kernel<<<cb_x, 256, 0, stream>>>(q,  Xq,  (int)NE);
  cvt_bf16_kernel<<<cb_x, 256, 0, stream>>>(k,  Xk,  (int)NE);
  cvt_bf16_kernel<<<cb_x, 256, 0, stream>>>(v,  Xv,  (int)NE);
  cvt_bf16_kernel<<<cb_w, 256, 0, stream>>>(Wq, Wqb, (int)NW);
  cvt_bf16_kernel<<<cb_w, 256, 0, stream>>>(Wk, Wkb, (int)NW);
  cvt_bf16_kernel<<<cb_w, 256, 0, stream>>>(Wv, Wvb, (int)NW);
  cvt_bf16_kernel<<<cb_w, 256, 0, stream>>>(Wo, Wob, (int)NW);

  dim3 gemm_grid(M_ / 64, D_ / 64);  // (64, 16)
  gemm_xwT_kernel<0><<<gemm_grid, 128, 0, stream>>>(Xq, Wqb, Qh, nullptr);
  gemm_xwT_kernel<0><<<gemm_grid, 128, 0, stream>>>(Xk, Wkb, Kh, nullptr);
  gemm_xwT_kernel<1><<<gemm_grid, 128, 0, stream>>>(Xv, Wvb, Vt, nullptr);

  dim3 attn_grid(S_ / 16 / 4, B_ * H_);  // (32, 32)
  flash_attn_kernel<<<attn_grid, 128, 0, stream>>>(Qh, Kh, Vt, Ob);

  gemm_xwT_kernel<2><<<gemm_grid, 128, 0, stream>>>(Ob, Wob, nullptr, out);
}
